// AnyAttention_30786325578183
// MI455X (gfx1250) — matrix operand order repaired and checked by reading, not compiled, not run on hardware
//
#include <hip/hip_runtime.h>
#include <hip/hip_bf16.h>
#include <stdint.h>

#define Nn 2048
#define Bb 2
#define Dd 256
#define Gg 8
#define Cc 32
#define NB (Nn * Bb) /* 4096 rows */

typedef __attribute__((ext_vector_type(16))) __bf16 v16bf;
typedef __attribute__((ext_vector_type(8)))  __bf16 v8bf;
typedef __attribute__((ext_vector_type(8)))  float  v8f;

static __device__ __forceinline__ v8f wmma_bf16(v16bf a, v16bf b, v8f c) {
  // D = A(16x32 bf16) * B(32x16 bf16) + C(16x16 f32)
  return __builtin_amdgcn_wmma_f32_16x16x32_bf16(false, a, false, b, (short)0, c,
                                                 false, false);
}

// gfx1250 async global->LDS copy (tracked by ASYNCcnt), 16 bytes per lane.
static __device__ __forceinline__ void async_load_b128(uint32_t lds_off,
                                                       const void* gaddr) {
  asm volatile("global_load_async_to_lds_b128 %0, %1, off"
               :: "v"(lds_off), "v"(gaddr) : "memory");
}
static __device__ __forceinline__ void wait_async0() {
  asm volatile("s_wait_asynccnt 0x0" ::: "memory");
}

static __device__ __forceinline__ float red_max16(float x) {
#pragma unroll
  for (int m = 1; m < 16; m <<= 1) x = fmaxf(x, __shfl_xor(x, m, 32));
  return x;
}
static __device__ __forceinline__ float red_sum16(float x) {
#pragma unroll
  for (int m = 1; m < 16; m <<= 1) x += __shfl_xor(x, m, 32);
  return x;
}
static __device__ __forceinline__ float red_sum32(float x) {
#pragma unroll
  for (int m = 1; m < 32; m <<= 1) x += __shfl_xor(x, m, 32);
  return x;
}

// ---------------- LayerNorm: one wave per (n,b) row, f32 -> bf16 -------------
__global__ void __launch_bounds__(256) ln_kernel(const float* __restrict__ x,
                                                 const float* __restrict__ w,
                                                 const float* __restrict__ bsft,
                                                 __bf16* __restrict__ y) {
  const int wave = threadIdx.x >> 5, lane = threadIdx.x & 31;
  const int row = blockIdx.x * 8 + wave;  // 512 blocks * 8 waves = 4096 rows
  const float* xr = x + (size_t)row * Dd;
  float v[8];
  float s = 0.f;
#pragma unroll
  for (int i = 0; i < 8; i++) { v[i] = xr[lane + 32 * i]; s += v[i]; }
  s = red_sum32(s);
  const float mean = s * (1.0f / Dd);
  float var = 0.f;
#pragma unroll
  for (int i = 0; i < 8; i++) { float d = v[i] - mean; var += d * d; }
  var = red_sum32(var) * (1.0f / Dd);
  const float rstd = rsqrtf(var + 1e-5f);
  __bf16* yr = y + (size_t)row * Dd;
#pragma unroll
  for (int i = 0; i < 8; i++) {
    const int c = lane + 32 * i;
    yr[c] = (__bf16)((v[i] - mean) * rstd * w[c] + bsft[c]);
  }
}

// ---------------- f32 -> bf16 convert (for weight matrices) ------------------
__global__ void f32_to_bf16_kernel(const float* __restrict__ s,
                                   __bf16* __restrict__ d, int n) {
  int i = blockIdx.x * blockDim.x + threadIdx.x;
  if (i < n) d[i] = (__bf16)s[i];
}

// ---------------- colsum2[d] = sum_e W2[e,d];  colsum2[D] = sum(b2) ----------
__global__ void __launch_bounds__(256) colsum_kernel(const float* __restrict__ W2,
                                                     const float* __restrict__ b2,
                                                     float* __restrict__ cs) {
  const int d = threadIdx.x;
  float s = 0.f;
  for (int e = 0; e < Dd; e++) s += W2[e * Dd + d];
  cs[d] = s;
  if (d == 0) {
    float sb = 0.f;
    for (int e = 0; e < Dd; e++) sb += b2[e];
    cs[Dd] = sb;
  }
}

// ---------------- Generic bf16 WMMA GEMM: Y[m,e] = sum_d X[m,d]*W[e,d] -------
// 4096x256 = X(4096x256) @ W(256x256)^T ; one wave per 16x16 tile, K-loop of 8.
__global__ void __launch_bounds__(256) gemm16_kernel(const __bf16* __restrict__ X,
                                                     const __bf16* __restrict__ W,
                                                     float* __restrict__ Yf,
                                                     __bf16* __restrict__ Ybf) {
  const int wave = threadIdx.x >> 5, lane = threadIdx.x & 31;
  const int tile = blockIdx.x * 8 + wave;  // 4096 tiles total (256 m x 16 e)
  const int m0 = (tile >> 4) << 4;
  const int e0 = (tile & 15) << 4;
  const int h = lane >> 4, l15 = lane & 15;
  const __bf16* xr = X + (size_t)(m0 + l15) * Dd;  // A: row M = lane&15
  const __bf16* wr = W + (size_t)(e0 + l15) * Dd;  // B: col N = lane&15 = W row
  v8f c = {0.f, 0.f, 0.f, 0.f, 0.f, 0.f, 0.f, 0.f};
#pragma unroll
  for (int k0 = 0; k0 < Dd; k0 += 32) {
    v16bf a = *(const v16bf*)(xr + k0 + 16 * h);
    v16bf b = *(const v16bf*)(wr + k0 + 16 * h);  // 16 contiguous K values
    c = wmma_bf16(a, b, c);
  }
#pragma unroll
  for (int v = 0; v < 8; v++) {
    const int m = m0 + v + 8 * h;   // C/D layout: VGPR v -> rows v / v+8
    const int e = e0 + l15;
    if (Yf)  Yf[(size_t)m * Dd + e] = c[v];
    if (Ybf) Ybf[(size_t)m * Dd + e] = (__bf16)c[v];
  }
}

// ---------------- Flash attention + fused t = out . colsum2 ------------------
// grid (N/64, G, B), block 128 (4 waves, each owns 16 q-rows).
// K chunk staged via async global->LDS DMA; V chunk staged transposed (d-major)
// via an in-register 8x8 subtile transpose so PV B operands are contiguous.
__global__ void __launch_bounds__(128)
attn_kernel(const __bf16* __restrict__ qp, const __bf16* __restrict__ kp,
            const __bf16* __restrict__ vp, const float* __restrict__ cs2,
            float* __restrict__ outw, float* __restrict__ tvals) {
  __shared__ __bf16 Kt[32 * Cc];        // K chunk, row-major (32 k x 32 c)
  __shared__ __bf16 VtT[Dd * 32];       // V chunk, TRANSPOSED (256 d x 32 k)
  __shared__ __bf16 Pr[4][16 * 32];     // per-wave probs tile (16 q x 32 k)

  const int tid = threadIdx.x;
  const int wave = tid >> 5, lane = tid & 31;
  const int h = lane >> 4, l15 = lane & 15;
  const int b = blockIdx.z, g = blockIdx.y;
  const int q0 = blockIdx.x * 64 + wave * 16;
  const float scale = 0.17677669529663687f;  // C^-0.5 = 1/sqrt(32)

  // Q operand in A layout (held for the whole kernel)
  const __bf16* qptr = qp + ((size_t)(q0 + l15) * Bb + b) * Dd + g * Cc;
  v16bf qa = *(const v16bf*)(qptr + 16 * h);

  v8f acc[16];  // 16 d-tiles x (16x16 f32) -> 16 q-rows x 256 d in registers
  const v8f zc = {0.f, 0.f, 0.f, 0.f, 0.f, 0.f, 0.f, 0.f};
#pragma unroll
  for (int t = 0; t < 16; t++) acc[t] = zc;
  float mrow[8], lsum[8];
#pragma unroll
  for (int v = 0; v < 8; v++) { mrow[v] = -3.0e38f; lsum[v] = 0.f; }

  // per-thread staging roles (fixed across the k loop)
  const int krr = tid >> 2;          // 0..31 : K row
  const int kcc = (tid & 3) * 8;     // 0/8/16/24 : K col (8 bf16 = 16B)
  const uint32_t kt_lds = (uint32_t)(uintptr_t)&Kt[krr * Cc + kcc];
  const int kk0 = (tid & 3) * 8;     // V subtile: k range
  const int d0  = (tid >> 2) * 8;    // V subtile: d range (32 threads * 8 = 256)

  for (int k0 = 0; k0 < Nn; k0 += 32) {
    __syncthreads();  // protect LDS K/V from previous iteration's readers

    // K chunk: async DMA, one b128 per lane (32x32 bf16 tile)
    async_load_b128(kt_lds,
                    kp + ((size_t)(k0 + krr) * Bb + b) * Dd + g * Cc + kcc);

    // V chunk: load 8x8 subtile, transpose in registers, store d-major
    {
      v8bf rows[8];
#pragma unroll
      for (int j = 0; j < 8; j++)
        rows[j] = *(const v8bf*)(vp + ((size_t)(k0 + kk0 + j) * Bb + b) * Dd + d0);
#pragma unroll
      for (int dd = 0; dd < 8; dd++) {
        v8bf col;
#pragma unroll
        for (int j = 0; j < 8; j++) col[j] = rows[j][dd];
        *(v8bf*)&VtT[(d0 + dd) * 32 + kk0] = col;
      }
    }
    wait_async0();    // this wave's async K copy is in LDS
    __syncthreads();  // publish K and V tiles to all waves

    // scores: S = Q(16x32) * K^T -> two 16x16 tiles over the 32-k chunk
    v16bf kb0 = *(const v16bf*)&Kt[(l15) * Cc + 16 * h];
    v16bf kb1 = *(const v16bf*)&Kt[(16 + l15) * Cc + 16 * h];
    v8f s0 = wmma_bf16(qa, kb0, zc);
    v8f s1 = wmma_bf16(qa, kb1, zc);

    // online softmax; rows live at (VGPR v, half h) -> 16-lane reductions
    float al[8];
#pragma unroll
    for (int v = 0; v < 8; v++) {
      const float x0 = s0[v] * scale, x1 = s1[v] * scale;
      const float m_new = fmaxf(mrow[v], red_max16(fmaxf(x0, x1)));
      const float a = __expf(mrow[v] - m_new);
      const float p0 = __expf(x0 - m_new);
      const float p1 = __expf(x1 - m_new);
      lsum[v] = lsum[v] * a + red_sum16(p0 + p1);
      mrow[v] = m_new;
      al[v] = a;
      // stash probs row-major: C-layout (M=v+8h, N=l15) -> LDS
      Pr[wave][(v + 8 * h) * 32 + l15] = (__bf16)p0;
      Pr[wave][(v + 8 * h) * 32 + 16 + l15] = (__bf16)p1;
    }
#pragma unroll
    for (int t = 0; t < 16; t++)
#pragma unroll
      for (int v = 0; v < 8; v++) acc[t][v] *= al[v];

    // reload probs in A layout (row = lane&15)
    v16bf pa = *(const v16bf*)&Pr[wave][l15 * 32 + 16 * h];

    // O += P(16x32) * V(32x256): 16 WMMAs; B operand = one contiguous 32B
    // slab of the transposed V tile (lane -> row d = t*16 + l15).
#pragma unroll
    for (int t = 0; t < 16; t++) {
      v16bf vb = *(const v16bf*)&VtT[(t * 16 + l15) * 32 + 16 * h];
      acc[t] = wmma_bf16(pa, vb, acc[t]);
    }
  }

  // epilogue: normalize, write out, fused t = out . colsum2 + sum(b2)
  float linv[8];
#pragma unroll
  for (int v = 0; v < 8; v++) linv[v] = 1.0f / lsum[v];
  float tp[8];
#pragma unroll
  for (int v = 0; v < 8; v++) tp[v] = 0.f;
#pragma unroll
  for (int t = 0; t < 16; t++) {
    const float csv = cs2[t * 16 + l15];
#pragma unroll
    for (int v = 0; v < 8; v++) {
      const float o = acc[t][v] * linv[v];
      tp[v] += o * csv;
      const int qn = q0 + v + 8 * h;
      outw[(((size_t)b * Nn + qn) * Gg + g) * Dd + t * 16 + l15] = o;
    }
  }
  const float sb2 = cs2[Dd];
#pragma unroll
  for (int v = 0; v < 8; v++) {
    const float tv = red_sum16(tp[v]) + sb2;
    if (l15 == 0) {
      const int qn = q0 + v + 8 * h;
      tvals[((size_t)b * Nn + qn) * Gg + g] = tv;
    }
  }
}

// ---------------- total[b,g] = sum_n t[b,n,g] --------------------------------
__global__ void __launch_bounds__(256) ttotal_kernel(const float* __restrict__ tvals,
                                                     float* __restrict__ total) {
  const int bg = blockIdx.x;
  const int b = bg / Gg, g = bg % Gg;
  float s = 0.f;
  for (int n = threadIdx.x; n < Nn; n += 256)
    s += tvals[((size_t)b * Nn + n) * Gg + g];
  __shared__ float sh[256];
  sh[threadIdx.x] = s;
  __syncthreads();
  for (int o = 128; o > 0; o >>= 1) {
    if (threadIdx.x < o) sh[threadIdx.x] += sh[threadIdx.x + o];
    __syncthreads();
  }
  if (threadIdx.x == 0) total[bg] = sh[0];
}

// ---------------- u[m,d] = sum_g out*off_diag ; osum[m,d] = sum_g out --------
__global__ void __launch_bounds__(256) uosum_kernel(const float* __restrict__ outw,
                                                    const float* __restrict__ tvals,
                                                    const float* __restrict__ total,
                                                    __bf16* __restrict__ u_bf,
                                                    float* __restrict__ osum) {
  const size_t idx = (size_t)blockIdx.x * 256 + threadIdx.x;  // NB*Dd
  const int d = (int)(idx & 255);
  const size_t m = idx >> 8;       // m = b*N + n
  const int b = (int)(m >> 11);    // m / 2048
  float u = 0.f, os = 0.f;
#pragma unroll
  for (int g = 0; g < Gg; g++) {
    const float off = total[b * Gg + g] - tvals[m * Gg + g];
    const float o = outw[(m * Gg + g) * Dd + d];
    u += o * off;
    os += o;
  }
  u_bf[idx] = (__bf16)u;
  osum[idx] = os;
}

// ---------------- gate + head-mean + transpose to (n,b,d) --------------------
__global__ void __launch_bounds__(256) final_kernel(const float* __restrict__ Agemm,
                                                    const float* __restrict__ osum,
                                                    const float* __restrict__ tvals,
                                                    const float* __restrict__ total,
                                                    const float* __restrict__ b1,
                                                    const float* __restrict__ mlp_w,
                                                    const float* __restrict__ mlp_b,
                                                    float* __restrict__ out) {
  const size_t idx = (size_t)blockIdx.x * 256 + threadIdx.x;
  const int d = (int)(idx & 255);
  const size_t m = idx >> 8;
  const int b = (int)(m >> 11);
  const int n = (int)(m & 2047);
  float ods = 0.f;
#pragma unroll
  for (int g = 0; g < Gg; g++) ods += total[b * Gg + g] - tvals[m * Gg + g];
  const float inv = 1.0f / ((float)(Nn - 1) * (float)Dd);
  const float A = (Agemm[idx] + b1[d] * ods) * inv;
  // sigmoid(-A*w + b) = 1 / (1 + exp(A*w - b))
  const float gate = 1.0f / (1.0f + __expf(A * mlp_w[d] - mlp_b[d]));
  out[((size_t)n * Bb + b) * Dd + d] = osum[idx] * (1.0f / Gg) * gate;
}

extern "C" void kernel_launch(void* const* d_in, const int* in_sizes, int n_in,
                              void* d_out, int out_size, void* d_ws, size_t ws_size,
                              hipStream_t stream) {
  (void)in_sizes; (void)n_in; (void)out_size; (void)ws_size;
  const float* q      = (const float*)d_in[0];
  const float* k      = (const float*)d_in[1];
  const float* v      = (const float*)d_in[2];
  const float* ln_q_w = (const float*)d_in[3];
  const float* ln_q_b = (const float*)d_in[4];
  const float* ln_k_w = (const float*)d_in[5];
  const float* ln_k_b = (const float*)d_in[6];
  const float* ln_v_w = (const float*)d_in[7];
  const float* ln_v_b = (const float*)d_in[8];
  const float* Wq     = (const float*)d_in[9];
  const float* Wk     = (const float*)d_in[10];
  const float* Wv     = (const float*)d_in[11];
  const float* W1     = (const float*)d_in[12];
  const float* b1     = (const float*)d_in[13];
  const float* W2     = (const float*)d_in[14];
  const float* b2     = (const float*)d_in[15];
  const float* mlp_w  = (const float*)d_in[16];
  const float* mlp_b  = (const float*)d_in[17];
  float* out = (float*)d_out;

  char* ws = (char*)d_ws;
  size_t off = 0;
  auto alloc = [&](size_t bytes) -> char* {
    char* p = ws + off;
    off += (bytes + 255) & ~(size_t)255;
    return p;
  };
  __bf16* xlnq = (__bf16*)alloc((size_t)NB * Dd * 2);
  __bf16* xlnk = (__bf16*)alloc((size_t)NB * Dd * 2);
  __bf16* xlnv = (__bf16*)alloc((size_t)NB * Dd * 2);
  __bf16* wq_bf = (__bf16*)alloc((size_t)Dd * Dd * 2);
  __bf16* wk_bf = (__bf16*)alloc((size_t)Dd * Dd * 2);
  __bf16* wv_bf = (__bf16*)alloc((size_t)Dd * Dd * 2);
  __bf16* w1_bf = (__bf16*)alloc((size_t)Dd * Dd * 2);
  __bf16* qpb  = (__bf16*)alloc((size_t)NB * Dd * 2);
  __bf16* kpb  = (__bf16*)alloc((size_t)NB * Dd * 2);
  __bf16* vpb  = (__bf16*)alloc((size_t)NB * Dd * 2);
  float* outw  = (float*)alloc((size_t)Bb * Nn * Gg * Dd * 4);  // 33.5 MB
  float* tvals = (float*)alloc((size_t)Bb * Nn * Gg * 4);
  float* total = (float*)alloc((size_t)Bb * Gg * 4);
  float* cs2   = (float*)alloc((size_t)(Dd + 1) * 4);
  __bf16* u_bf = (__bf16*)alloc((size_t)NB * Dd * 2);
  float* osum  = (float*)alloc((size_t)NB * Dd * 4);
  float* Agemm = (float*)alloc((size_t)NB * Dd * 4);

  // stage 0: weight converts, colsum, layernorms
  f32_to_bf16_kernel<<<256, 256, 0, stream>>>(Wq, wq_bf, Dd * Dd);
  f32_to_bf16_kernel<<<256, 256, 0, stream>>>(Wk, wk_bf, Dd * Dd);
  f32_to_bf16_kernel<<<256, 256, 0, stream>>>(Wv, wv_bf, Dd * Dd);
  f32_to_bf16_kernel<<<256, 256, 0, stream>>>(W1, w1_bf, Dd * Dd);
  colsum_kernel<<<1, 256, 0, stream>>>(W2, b2, cs2);
  ln_kernel<<<512, 256, 0, stream>>>(q, ln_q_w, ln_q_b, xlnq);
  ln_kernel<<<512, 256, 0, stream>>>(k, ln_k_w, ln_k_b, xlnk);
  ln_kernel<<<512, 256, 0, stream>>>(v, ln_v_w, ln_v_b, xlnv);

  // stage 1: QKV projections (WMMA)
  gemm16_kernel<<<512, 256, 0, stream>>>(xlnq, wq_bf, nullptr, qpb);
  gemm16_kernel<<<512, 256, 0, stream>>>(xlnk, wk_bf, nullptr, kpb);
  gemm16_kernel<<<512, 256, 0, stream>>>(xlnv, wv_bf, nullptr, vpb);

  // stage 2: flash attention (WMMA + async LDS staging) + fused t
  attn_kernel<<<dim3(Nn / 64, Gg, Bb), 128, 0, stream>>>(qpb, kpb, vpb, cs2,
                                                         outw, tvals);

  // stage 3: gate pipeline
  ttotal_kernel<<<Bb * Gg, 256, 0, stream>>>(tvals, total);
  uosum_kernel<<<NB * Dd / 256, 256, 0, stream>>>(outw, tvals, total, u_bf, osum);
  gemm16_kernel<<<512, 256, 0, stream>>>(u_bf, w1_bf, Agemm, nullptr);  // A = u @ W1^T
  final_kernel<<<NB * Dd / 256, 256, 0, stream>>>(Agemm, osum, tvals, total, b1,
                                                  mlp_w, mlp_b, out);
}
